// Box2Condition_67061619360230
// MI455X (gfx1250) — matrix operand order repaired and checked by reading, not compiled, not run on hardware
//
#include <hip/hip_runtime.h>
#include <hip/hip_bf16.h>

// ---------------------------------------------------------------------------
// MI455X (gfx1250, wave32) implementation.
// All contractions run on v_wmma_f32_16x16x32_f16.
// Activations are f16 channel-last so each lane's WMMA B-fragment is one
// contiguous 32B load; weights are pre-packed f16 [M][K] so the A-fragment is
// two contiguous 16B loads per lane (same contiguous k = 16h + s order in A and B).
// The deformable-conv kernel stages its weight K-chunks into LDS with the
// Tensor Data Mover (tensor_load_to_lds, TENSORcnt), double-buffered so the
// DMA overlaps the bilinear-gather + WMMA compute.
// ---------------------------------------------------------------------------

typedef __attribute__((ext_vector_type(16))) _Float16 hvec16;
typedef __attribute__((ext_vector_type(8)))  _Float16 hvec8;
typedef __attribute__((ext_vector_type(8)))  float    fvec8;
typedef __attribute__((ext_vector_type(4)))  unsigned int u32x4;
typedef __attribute__((ext_vector_type(8)))  int i32x8;
typedef __attribute__((ext_vector_type(4)))  int i32x4;

#define Bb   2
#define Nn   16384
#define Cin  32
#define Hh   64
#define Ww   1024
#define Pp   (Hh * Ww)      // 65536
#define Cmid 128
#define Cout 128
#define OH   128
#define OW   2048
#define WROW 40             // padded LDS row stride in halves (80 B)

__device__ __forceinline__ fvec8 zero8() {
  fvec8 z;
#pragma unroll
  for (int i = 0; i < 8; ++i) z[i] = 0.f;
  return z;
}
__device__ __forceinline__ hvec16 zero16() {
  hvec16 z;
#pragma unroll
  for (int i = 0; i < 16; ++i) z[i] = (_Float16)0.f;
  return z;
}

__device__ __forceinline__ fvec8 wmma_f16(hvec16 a, hvec16 b, fvec8 c) {
  // (neg_a, A, neg_b, B, c_mod, C, reuse_a, reuse_b)
  return __builtin_amdgcn_wmma_f32_16x16x32_f16(false, a, false, b, (short)0, c,
                                                false, false);
}

// A-fragment (16x32 f16, M x K). Per ISA: lane m = lane&15, half h = lane>>4.
// Frag halves 0..15 = K (16h .. 16h+15) in this code's contiguous order, shared with B:
// two contiguous v8h loads from row-major [M][K] storage.
// `base` must already point at (row, kc0) of the weight matrix.
__device__ __forceinline__ hvec16 load_a_frag(const _Float16* base) {
  int h = (threadIdx.x & 31) >> 4;
  hvec8 lo = *(const hvec8*)(base + 16 * h);
  hvec8 hi = *(const hvec8*)(base + 16 * h + 8);
  hvec16 r;
#pragma unroll
  for (int j = 0; j < 8; ++j) { r[j] = lo[j]; r[j + 8] = hi[j]; }
  return r;
}

// ---------------------------------------------------------------------------
// Tensor Data Mover: DMA one [128 rows x 32 halves] weight chunk (row stride
// 1152 halves in global) into LDS at lds_off, inserting 16B of padding after
// every 64B so LDS rows land at 80B (WROW) stride.  D# per ISA 8.3/8.4:
//   group0: count=1, lds_addr, 57-bit global_addr, type=2
//   group1: data_size=2B, pad_enable, pad_interval=16DW, pad_amount=4DW,
//           tensor_dim0=1152, tensor_dim1=128, tile_dim0=32, tile_dim1=128,
//           tensor_dim0_stride=1152
// ---------------------------------------------------------------------------
__device__ __forceinline__ void tdm_load_wchunk(const _Float16* gsrc,
                                                unsigned lds_off) {
  unsigned long long ga = (unsigned long long)(const void*)gsrc;
  u32x4 g0;
  g0[0] = 1u;                                   // count = 1 (valid descriptor)
  g0[1] = lds_off;                              // lds_addr (bytes)
  g0[2] = (unsigned)ga;                         // global_addr[31:0]
  g0[3] = (unsigned)(ga >> 32) | (2u << 30);    // global_addr[56:32] | type=2
  i32x8 g1;
  g1[0] = (1 << 16)                             // data_size = 2 bytes
        | (1 << 20)                             // pad_enable
        | (3 << 22)                             // pad_interval: 16 DWORDs
        | (3 << 25);                            // pad_amount: 4 DWORDs
  g1[1] = (int)(1152u << 16);                   // tensor_dim0 (bits 79:48)
  g1[2] = (int)(128u << 16);                    // tensor_dim1 (bits 111:80)
  g1[3] = (int)(32u << 16);                     // tile_dim0   (bits 127:112)
  g1[4] = 128;                                  // tile_dim1
  g1[5] = 1152;                                 // tensor_dim0_stride
  g1[6] = 0;
  g1[7] = 0;
  i32x4 z4;
  z4[0] = z4[1] = z4[2] = z4[3] = 0;
#if defined(__clang_major__) && (__clang_major__ >= 23)
  i32x8 z8;
#pragma unroll
  for (int j = 0; j < 8; ++j) z8[j] = 0;
  __builtin_amdgcn_tensor_load_to_lds(g0, g1, z4, z4, z8, 0);
#else
  __builtin_amdgcn_tensor_load_to_lds(g0, g1, z4, z4, 0);
#endif
}

// ---------------------------------------------------------------------------
// Stage 1: last-write scatter of point features into the range image
// ---------------------------------------------------------------------------
__global__ void k_winner_init(unsigned* winner) {
  int i = blockIdx.x * blockDim.x + threadIdx.x;
  if (i < Bb * Pp) winner[i] = 0u;
}

__global__ void k_scatter(const float* __restrict__ boxes, unsigned* winner) {
  int idx = blockIdx.x * blockDim.x + threadIdx.x;
  if (idx >= Bb * Nn) return;
  int b = idx / Nn, n = idx % Nn;
  const float* bx = boxes + ((size_t)b * Nn + n) * 6;
  float X = bx[0] * 80.f - 40.f;
  float Y = bx[1] * 80.f - 40.f;
  float Z = bx[2] * 4.f - 3.f;
  float depth = sqrtf(X * X + Y * Y + Z * Z) + 1e-8f;
  float yaw = -atan2f(Y, X);
  float s = fminf(1.f, fmaxf(-1.f, Z / depth));
  float pitch = asinf(s);
  const float PI = 3.14159265358979323846f;
  const float fd = 25.f * PI / 180.f;
  const float fov = 28.f * PI / 180.f;
  float u = 0.5f * (yaw / PI + 1.f) * (float)Ww;
  float v = (1.f - (pitch + fd) / fov) * (float)Hh;
  int xi = (int)floorf(u); xi = min(Ww - 1, max(0, xi));
  int yi = (int)floorf(v); yi = min(Hh - 1, max(0, yi));
  atomicMax(&winner[(size_t)b * Pp + yi * Ww + xi], (unsigned)(n + 1));
}

// Builds channel-last f16 feature map, K padded 39 -> 64.
__global__ void k_build_fmap(const unsigned* __restrict__ winner,
                             const float* __restrict__ feats,
                             const float* __restrict__ boxes,
                             const float* __restrict__ angles,
                             _Float16* __restrict__ fmap16) {
  int idx = blockIdx.x * blockDim.x + threadIdx.x;
  if (idx >= Bb * Pp) return;
  int b = idx / Pp;
  _Float16* dst = fmap16 + (size_t)idx * 64;
  unsigned w = winner[idx];
  if (w == 0u) {
#pragma unroll 4
    for (int c = 0; c < 64; ++c) dst[c] = (_Float16)0.f;
    return;
  }
  int n = (int)w - 1;
  const float* f = feats + ((size_t)b * Nn + n) * Cin;
#pragma unroll 4
  for (int c = 0; c < Cin; ++c) dst[c] = (_Float16)f[c];
  const float* bx = boxes + ((size_t)b * Nn + n) * 6;
#pragma unroll
  for (int j = 0; j < 6; ++j) dst[Cin + j] = (_Float16)bx[j];
  dst[38] = (_Float16)angles[(size_t)b * Nn + n];
#pragma unroll
  for (int c = 39; c < 64; ++c) dst[c] = (_Float16)0.f;
}

// ---------------------------------------------------------------------------
// Weight repacking to f16 [M][K]
// ---------------------------------------------------------------------------
__global__ void k_prep_wred(const float* __restrict__ w, _Float16* __restrict__ o) {
  int i = blockIdx.x * blockDim.x + threadIdx.x;
  if (i >= 128 * 64) return;
  int m = i >> 6, k = i & 63;
  o[i] = (k < 39) ? (_Float16)w[m * 39 + k] : (_Float16)0.f;
}

// w: [Mreal][128][3][3] -> o: [Mpad][1152], K = tap*128 + c
__global__ void k_prep_conv(const float* __restrict__ w, _Float16* __restrict__ o,
                            int Mreal, int Mpad) {
  int i = blockIdx.x * blockDim.x + threadIdx.x;
  if (i >= Mpad * 1152) return;
  int m = i / 1152, K = i % 1152;
  int k = K >> 7, c = K & 127;
  o[i] = (m < Mreal) ? (_Float16)w[(size_t)(m * 128 + c) * 9 + k] : (_Float16)0.f;
}

// w_up: [in=128][out=128][4][4] -> o: [4 parity classes][128 out][512], K = tap*128 + in
__global__ void k_prep_wup(const float* __restrict__ w, _Float16* __restrict__ o) {
  int i = blockIdx.x * blockDim.x + threadIdx.x;
  if (i >= 4 * 128 * 512) return;
  int cls = i >> 16;
  int rem = i & 65535;
  int m = rem >> 9;
  int K = rem & 511;
  int t = K >> 7, ic = K & 127;
  int py = cls >> 1, px = cls & 1;
  int ty = t >> 1, tx = t & 1;
  int ky = (py ^ 1) + 2 * ty, kx = (px ^ 1) + 2 * tx;
  o[i] = (_Float16)w[(size_t)((ic * 128 + m) * 4 + ky) * 4 + kx];
}

// ---------------------------------------------------------------------------
// Stage 2: 1x1 reduce 39->128 via WMMA. One wave = 16 pixels, all 8 M-chunks.
// ---------------------------------------------------------------------------
__global__ void __launch_bounds__(128)
k_reduce(const _Float16* __restrict__ fmap16, const _Float16* __restrict__ wred16,
         const float* __restrict__ bias, _Float16* __restrict__ x16) {
  int wave = threadIdx.x >> 5, lane = threadIdx.x & 31;
  int tile = blockIdx.x * 4 + wave;              // Bb*Pp/16 tiles
  int b = tile / (Pp / 16);
  int p0 = (tile % (Pp / 16)) * 16;
  int n = lane & 15, h = lane >> 4;
  size_t pix = (size_t)b * Pp + p0 + n;
  hvec16 b0 = *(const hvec16*)(fmap16 + pix * 64 + 16 * h);
  hvec16 b1 = *(const hvec16*)(fmap16 + pix * 64 + 32 + 16 * h);
#pragma unroll
  for (int mi = 0; mi < 8; ++mi) {
    const _Float16* wr = wred16 + (mi * 16 + n) * 64;
    hvec16 a0 = load_a_frag(wr);
    hvec16 a1 = load_a_frag(wr + 32);
    fvec8 acc = zero8();
    acc = wmma_f16(a0, b0, acc);
    acc = wmma_f16(a1, b1, acc);
    int ch0 = mi * 16 + 8 * h;
    hvec8 ov;
#pragma unroll
    for (int r = 0; r < 8; ++r) ov[r] = (_Float16)(acc[r] + bias[ch0 + r]);
    *(hvec8*)(x16 + pix * 128 + ch0) = ov;
  }
}

// ---------------------------------------------------------------------------
// Stage 3: 3x3 offset conv 128->18 (padded to 32 rows) via WMMA, K=1152.
// ---------------------------------------------------------------------------
__global__ void __launch_bounds__(128)
k_offconv(const _Float16* __restrict__ x16, const _Float16* __restrict__ woff16,
          const float* __restrict__ boff, float* __restrict__ offs) {
  int wave = threadIdx.x >> 5, lane = threadIdx.x & 31;
  int tile = blockIdx.x * 4 + wave;
  int b = tile / (Pp / 16);
  int p0 = (tile % (Pp / 16)) * 16;
  int y = p0 >> 10;
  int xbase = p0 & (Ww - 1);
  int n = lane & 15, h = lane >> 4;
  int xc = xbase + n;
  const _Float16* xb = x16 + (size_t)b * Pp * 128;
  fvec8 acc0 = zero8(), acc1 = zero8();
#pragma unroll 4
  for (int kc = 0; kc < 36; ++kc) {
    int k = kc >> 2, c0 = (kc & 3) * 32;
    int yy = y + k / 3 - 1, xx = xc + k % 3 - 1;
    hvec16 bf;
    if (yy >= 0 && yy < Hh && xx >= 0 && xx < Ww)
      bf = *(const hvec16*)(xb + (size_t)(yy * Ww + xx) * 128 + c0 + 16 * h);
    else
      bf = zero16();
    hvec16 a0 = load_a_frag(woff16 + (size_t)(0 * 16 + n) * 1152 + kc * 32);
    hvec16 a1 = load_a_frag(woff16 + (size_t)(1 * 16 + n) * 1152 + kc * 32);
    acc0 = wmma_f16(a0, bf, acc0);
    acc1 = wmma_f16(a1, bf, acc1);
  }
  size_t p = (size_t)p0 + n;
#pragma unroll
  for (int r = 0; r < 8; ++r) {
    int m = r + 8 * h;                                   // 0..15
    offs[((size_t)b * 18 + m) * Pp + p] = acc0[r] + boff[m];
    int m2 = 16 + m;                                     // 16..31, keep <18
    if (m2 < 18) offs[((size_t)b * 18 + m2) * Pp + p] = acc1[r] + boff[m2];
  }
}

// ---------------------------------------------------------------------------
// Stage 4: deformable 3x3 conv 128->128. Bilinear gather (packed f16 FMA)
// feeds 8 WMMA accumulators per wave. The [128x32] weight K-chunk is DMA'd
// into LDS by the Tensor Data Mover, double-buffered: wave 0 issues the TDM
// descriptor for chunk kc+1 while all waves compute chunk kc, then
// s_wait_tensorcnt + barrier.
// ---------------------------------------------------------------------------
__global__ void __launch_bounds__(128)
k_deform(const _Float16* __restrict__ x16, const _Float16* __restrict__ wdef16,
         const float* __restrict__ offs, const float* __restrict__ bdef,
         _Float16* __restrict__ xd16) {
  __shared__ _Float16 sW[2][128 * WROW];                 // 2 x 10 KB, 80B rows
  int wave = threadIdx.x >> 5, lane = threadIdx.x & 31;
  int tile = blockIdx.x * 4 + wave;
  int b = tile / (Pp / 16);
  int p0 = (tile % (Pp / 16)) * 16;
  int y = p0 >> 10;
  int xbase = p0 & (Ww - 1);
  int n = lane & 15, h = lane >> 4;
  int xc = xbase + n;
  size_t p = (size_t)p0 + n;
  const _Float16* xb = x16 + (size_t)b * Pp * 128;
  const float* ob = offs + (size_t)b * 18 * Pp;
  unsigned lds0 = (unsigned)(unsigned long long)(const void*)&sW[0][0];
  unsigned lds1 = (unsigned)(unsigned long long)(const void*)&sW[1][0];

  fvec8 acc[8];
#pragma unroll
  for (int mi = 0; mi < 8; ++mi) acc[mi] = zero8();

  // Prologue: DMA chunk 0 into buffer 0.
  if (threadIdx.x < 32) {
    tdm_load_wchunk(wdef16, lds0);
    __builtin_amdgcn_s_wait_tensorcnt(0);
  }
  __syncthreads();

  for (int kc = 0; kc < 36; ++kc) {
    // Kick DMA for the next chunk into the other buffer (overlaps compute).
    if (kc + 1 < 36 && threadIdx.x < 32)
      tdm_load_wchunk(wdef16 + (kc + 1) * 32, ((kc + 1) & 1) ? lds1 : lds0);

    int k = kc >> 2, c0 = (kc & 3) * 32;
    float offy = ob[(size_t)(2 * k) * Pp + p];
    float offx = ob[(size_t)(2 * k + 1) * Pp + p];
    float sy = (float)(y + k / 3 - 1) + offy;
    float sx = (float)(xc + k % 3 - 1) + offx;
    float fy0 = floorf(sy), fx0 = floorf(sx);
    float wy = sy - fy0, wx = sx - fx0;
    int y0 = (int)fy0, x0i = (int)fx0;

    hvec16 bf = zero16();
#pragma unroll
    for (int cy = 0; cy < 2; ++cy)
#pragma unroll
      for (int cx = 0; cx < 2; ++cx) {
        int yy = y0 + cy, xx = x0i + cx;
        float wgt = (cy ? wy : 1.f - wy) * (cx ? wx : 1.f - wx);
        if (yy >= 0 && yy < Hh && xx >= 0 && xx < Ww) {
          hvec16 v = *(const hvec16*)(xb + (size_t)(yy * Ww + xx) * 128 + c0 + 16 * h);
          bf += v * (_Float16)wgt;                       // v_pk_fma_f16
        }
      }

    const _Float16* wbuf = sW[kc & 1];
#pragma unroll
    for (int mi = 0; mi < 8; ++mi) {
      hvec16 a = load_a_frag(wbuf + (mi * 16 + n) * WROW);
      acc[mi] = wmma_f16(a, bf, acc[mi]);
    }

    // Wave 0 drains its TDM op; barrier publishes the new chunk to the block
    // and guarantees everyone is done reading before the buffer is reused.
    if (threadIdx.x < 32) __builtin_amdgcn_s_wait_tensorcnt(0);
    __syncthreads();
  }

#pragma unroll
  for (int mi = 0; mi < 8; ++mi) {
    int ch0 = mi * 16 + 8 * h;
    hvec8 ov;
#pragma unroll
    for (int r = 0; r < 8; ++r) ov[r] = (_Float16)(acc[mi][r] + bdef[ch0 + r]);
    *(hvec8*)(xd16 + ((size_t)b * Pp + p) * 128 + ch0) = ov;
  }
}

// ---------------------------------------------------------------------------
// Stage 5: ConvTranspose2d k=4 s=2 p=1 as 4 same-parity GEMMs (K=512).
// One wave = 16 out-ch x 16 same-x-parity output pixels of one row.
// ---------------------------------------------------------------------------
__global__ void __launch_bounds__(128)
k_upconv(const _Float16* __restrict__ xd16, const _Float16* __restrict__ wup16,
         const float* __restrict__ bup, float* __restrict__ out) {
  int wave = threadIdx.x >> 5, lane = threadIdx.x & 31;
  int w = blockIdx.x * 4 + wave;     // Bb*8*128*128 waves
  int xt = w & 127; w >>= 7;
  int oy = w & 127; w >>= 7;
  int mi = w & 7;
  int b = w >> 3;
  int px = xt & 1;
  int xb0 = (xt >> 1) * 32;
  int n = lane & 15, h = lane >> 4;
  int ox = xb0 + px + 2 * n;
  int py = oy & 1;
  int cls = py * 2 + px;
  const _Float16* xdb = xd16 + (size_t)b * Pp * 128;
  const _Float16* wb = wup16 + (size_t)cls * 128 * 512;
  fvec8 acc = zero8();
#pragma unroll 4
  for (int kc = 0; kc < 16; ++kc) {
    int t = kc >> 2, i0 = (kc & 3) * 32;
    int ty = t >> 1, tx = t & 1;
    int ky = (py ^ 1) + 2 * ty, kx = (px ^ 1) + 2 * tx;
    int numy = oy + 1 - ky, iy = numy >> 1;
    int numx = ox + 1 - kx, ix = numx >> 1;
    hvec16 bf;
    if (numy >= 0 && iy < Hh && numx >= 0 && ix < Ww)
      bf = *(const hvec16*)(xdb + (size_t)(iy * Ww + ix) * 128 + i0 + 16 * h);
    else
      bf = zero16();
    hvec16 a = load_a_frag(wb + (size_t)(mi * 16 + n) * 512 + kc * 32);
    acc = wmma_f16(a, bf, acc);
  }
#pragma unroll
  for (int r = 0; r < 8; ++r) {
    int o = mi * 16 + r + 8 * h;
    out[(((size_t)b * Cout + o) * OH + oy) * OW + ox] = acc[r] + bup[o];
  }
}

// ---------------------------------------------------------------------------
extern "C" void kernel_launch(void* const* d_in, const int* in_sizes, int n_in,
                              void* d_out, int out_size, void* d_ws, size_t ws_size,
                              hipStream_t stream) {
  (void)in_sizes; (void)n_in; (void)out_size; (void)ws_size;
  const float* dec_boxes  = (const float*)d_in[0];
  const float* dec_angles = (const float*)d_in[1];
  const float* features   = (const float*)d_in[2];
  const float* w_reduce   = (const float*)d_in[3];
  const float* b_reduce   = (const float*)d_in[4];
  const float* w_off      = (const float*)d_in[5];
  const float* b_off      = (const float*)d_in[6];
  const float* w_def      = (const float*)d_in[7];
  const float* b_def      = (const float*)d_in[8];
  const float* w_up       = (const float*)d_in[9];
  const float* b_up       = (const float*)d_in[10];
  float* out = (float*)d_out;

  // Workspace carve-out (~91 MB total).
  char* ws = (char*)d_ws;
  auto take = [&](size_t bytes) {
    char* r = ws;
    ws += (bytes + 255) & ~(size_t)255;
    return r;
  };
  unsigned* winner = (unsigned*)take((size_t)Bb * Pp * 4);
  _Float16* fmap16 = (_Float16*)take((size_t)Bb * Pp * 64 * 2);
  _Float16* x16    = (_Float16*)take((size_t)Bb * Pp * 128 * 2);
  float*    offs   = (float*)take((size_t)Bb * 18 * Pp * 4);
  _Float16* xd16   = (_Float16*)take((size_t)Bb * Pp * 128 * 2);
  _Float16* wred16 = (_Float16*)take((size_t)128 * 64 * 2);
  _Float16* woff16 = (_Float16*)take((size_t)32 * 1152 * 2);
  _Float16* wdef16 = (_Float16*)take((size_t)128 * 1152 * 2);
  _Float16* wup16  = (_Float16*)take((size_t)4 * 128 * 512 * 2);

  k_winner_init<<<(Bb * Pp + 255) / 256, 256, 0, stream>>>(winner);
  k_scatter<<<(Bb * Nn + 255) / 256, 256, 0, stream>>>(dec_boxes, winner);
  k_build_fmap<<<(Bb * Pp + 255) / 256, 256, 0, stream>>>(winner, features, dec_boxes,
                                                          dec_angles, fmap16);
  k_prep_wred<<<(128 * 64 + 255) / 256, 256, 0, stream>>>(w_reduce, wred16);
  k_prep_conv<<<(32 * 1152 + 255) / 256, 256, 0, stream>>>(w_off, woff16, 18, 32);
  k_prep_conv<<<(128 * 1152 + 255) / 256, 256, 0, stream>>>(w_def, wdef16, 128, 128);
  k_prep_wup<<<(4 * 128 * 512 + 255) / 256, 256, 0, stream>>>(w_up, wup16);

  int tiles = Bb * (Pp / 16);                 // 8192 wave-tiles
  k_reduce<<<tiles / 4, 128, 0, stream>>>(fmap16, wred16, b_reduce, x16);
  k_offconv<<<tiles / 4, 128, 0, stream>>>(x16, woff16, b_off, offs);
  k_deform<<<tiles / 4, 128, 0, stream>>>(x16, wdef16, offs, b_def, xd16);

  int upwaves = Bb * 8 * OH * 128;            // 262144 wave-tiles
  k_upconv<<<upwaves / 4, 128, 0, stream>>>(xd16, wup16, b_up, out);
}
